// IoU_44796508897918
// MI455X (gfx1250) — compile-verified, matrix operand order checked by reading, not run on hardware
//
#include <hip/hip_runtime.h>

typedef __attribute__((ext_vector_type(4))) float v4f;
typedef __attribute__((ext_vector_type(8))) int   v8i;

#define PLANES           192           // B*C = 64*3
#define PLANE_PIX        (512*512)     // 262144
#define BLOCKS_PER_PLANE 8
#define CHUNK_PIX        (PLANE_PIX / BLOCKS_PER_PLANE)   // 32768
#define THREADS          256
#define ITERS            (CHUNK_PIX / (THREADS * 4))      // 32 float4 iters/thread

// ---------------------------------------------------------------------------
// Kernel 0: zero the per-plane accumulators (re-run every launch: graph-safe)
// ---------------------------------------------------------------------------
__global__ void iou_zero(unsigned int* __restrict__ acc) {
    for (int i = threadIdx.x; i < PLANES * 3; i += blockDim.x) acc[i] = 0u;
}

// ---------------------------------------------------------------------------
// Kernel 1: streaming threshold + count. Per-thread counts packed into bytes
// (inter | union<<8 | tgt<<16, each <=128). Block reduction via WMMA IU8:
// 256 packed u32s = 1024 bytes = one 16x64 IU8 A-matrix; B = all-ones bytes
// => every D element is a row-sum. Sum 8 D VGPRs + cross-half shuffle = total.
// ---------------------------------------------------------------------------
__global__ __launch_bounds__(THREADS)
void iou_main(const float* __restrict__ inp, const float* __restrict__ tgt,
              unsigned int* __restrict__ acc) {
    __shared__ unsigned int lds[THREADS];

    const int  tid   = threadIdx.x;
    const int  plane = blockIdx.x >> 3;     // 0..191
    const int  chunk = blockIdx.x & 7;      // 0..7
    const long base4 = (((long)plane * PLANE_PIX) + (long)chunk * CHUNK_PIX) >> 2;

    const v4f* __restrict__ in4 = (const v4f*)inp;
    const v4f* __restrict__ tg4 = (const v4f*)tgt;

    unsigned int packed = 0u;
    #pragma unroll 4
    for (int i = 0; i < ITERS; ++i) {
        const long idx = base4 + (long)i * THREADS + tid;   // coalesced b128
        v4f a = __builtin_nontemporal_load(&in4[idx]);      // TH=NT: stream past caches
        v4f b = __builtin_nontemporal_load(&tg4[idx]);
        #pragma unroll
        for (int j = 0; j < 4; ++j) {
            unsigned int ib = a[j] > 0.5f;
            unsigned int tb = b[j] > 0.5f;
            packed += (ib & tb) | ((ib | tb) << 8) | (tb << 16);
        }
    }
    lds[tid] = packed;
    __syncthreads();

    if (tid < 32) {                 // wave 0 only: EXEC all-ones for WMMA
        const int lane = tid;
        v8i bmat, cz = {};
        #pragma unroll
        for (int v = 0; v < 8; ++v) bmat[v] = 0x01010101;   // B = ones (layout-free)

        unsigned int totals[3];
        #pragma unroll
        for (int q = 0; q < 3; ++q) {                       // inter, union, tgt
            v8i amat;
            #pragma unroll
            for (int v = 0; v < 8; ++v)
                amat[v] = (int)((lds[v * 32 + lane] >> (8 * q)) & 0xFFu);
            // D = A(16x64 u8) * ones(64x16) + 0  -> D[m][n] = rowsum(m)
            v8i d = __builtin_amdgcn_wmma_i32_16x16x64_iu8(
                        /*sgn_a=*/false, amat, /*sgn_b=*/false, bmat,
                        cz, /*reuse_a=*/false, /*reuse_b=*/false);
            int partial = d[0] + d[1] + d[2] + d[3] + d[4] + d[5] + d[6] + d[7];
            partial += __shfl_down(partial, 16, 32);        // combine M halves
            totals[q] = (unsigned int)partial;              // valid in lane 0
        }
        if (lane == 0) {
            atomicAdd(&acc[plane * 3 + 0], totals[0]);      // integer atomics:
            atomicAdd(&acc[plane * 3 + 1], totals[1]);      // deterministic
            atomicAdd(&acc[plane * 3 + 2], totals[2]);
        }
    }
}

// ---------------------------------------------------------------------------
// Kernel 2: per-channel mean over batch with valid/union guards.
// 3 threads, each reduces its channel serially -> bit-deterministic.
// ---------------------------------------------------------------------------
__global__ void iou_final(const unsigned int* __restrict__ acc,
                          float* __restrict__ out) {
    const int c = threadIdx.x;
    if (c < 3) {
        float iouSum = 0.0f, cnt = 0.0f;
        for (int b = 0; b < 64; ++b) {
            const unsigned int inter = acc[(b * 3 + c) * 3 + 0];
            const unsigned int uni   = acc[(b * 3 + c) * 3 + 1];
            const unsigned int tg    = acc[(b * 3 + c) * 3 + 2];
            if (tg > 0u) {                                   // valid = tgt_sum > 0
                const float iou = (uni > 0u) ? ((float)inter / (float)uni) : 0.0f;
                iouSum += iou;
                cnt    += 1.0f;
            }
        }
        out[c] = (cnt > 0.0f) ? (iouSum / cnt) : 0.0f;
    }
}

extern "C" void kernel_launch(void* const* d_in, const int* in_sizes, int n_in,
                              void* d_out, int out_size, void* d_ws, size_t ws_size,
                              hipStream_t stream) {
    const float* inp = (const float*)d_in[0];
    const float* tgt = (const float*)d_in[1];
    float*       out = (float*)d_out;
    unsigned int* acc = (unsigned int*)d_ws;   // 192*3 u32 = 2304 B scratch

    iou_zero <<<1, 256, 0, stream>>>(acc);
    iou_main <<<PLANES * BLOCKS_PER_PLANE, THREADS, 0, stream>>>(inp, tgt, acc);
    iou_final<<<1, 32, 0, stream>>>(acc, out);
}
